// VQVAE_48558900249078
// MI455X (gfx1250) — matrix operand order repaired and checked by reading, not compiled, not run on hardware
//
#include <hip/hip_runtime.h>
#include <hip/hip_bf16.h>

typedef __attribute__((ext_vector_type(16))) __bf16 v16bf;
typedef __attribute__((ext_vector_type(8)))  __bf16 v8bf;
typedef __attribute__((ext_vector_type(8)))  float  v8f;

#define B_   2048
#define T_   64
#define H_   256
#define E_   64
#define V_   51
#define Z_   32
#define KCB_ 1000

#define PRED_OFF 2
#define KL_OFF   (2 + B_*(T_-1))   /* 129026 */
#define Q_OFF    (KL_OFF + 1)
#define IDX_OFF  (Q_OFF + 1)

// ---------------------------------------------------------------- WMMA utils

__device__ __forceinline__ v8f wmma_bf16(v16bf a, v16bf b, v8f c) {
  // D = A(16x32 bf16) * B(32x16 bf16) + C(16x16 f32)
  return __builtin_amdgcn_wmma_f32_16x16x32_bf16(
      /*neg_a=*/false, a, /*neg_b=*/false, b,
      /*c_mod=*/(short)0, c, /*reuse_a=*/false, /*reuse_b=*/false);
}

// A-fragment: lane needs bf16 elements [s..s+7] and [s+16..s+23] of its row.
// Two 16B loads, no conversion VALU.
__device__ __forceinline__ v16bf loadA(const __bf16* __restrict__ p) {
  v8bf lo = *(const v8bf*)(p);
  v8bf hi = *(const v8bf*)(p + 16);
  return __builtin_shufflevector(lo, hi, 0, 1, 2, 3, 4, 5, 6, 7,
                                 8, 9, 10, 11, 12, 13, 14, 15);
}

__device__ __forceinline__ float sig_(float x) { return 1.f / (1.f + __expf(-x)); }

// ------------------------------------------------------- weight pack kernels
// B-fragment layout (16x16x32 bf16): element j of lane L:
//   n = nTile*16 + (L & 15),  k = kChunk*32 + 8*(L>=16) + (j & 7) + 16*(j >= 8)
// Packed as P[((nTile*KC + kc)*32 + lane)*16 + j]  -> lane reads 32B contiguous.

__global__ void pack_b2(const float* __restrict__ Wa, int Ka,
                        const float* __restrict__ Wb, int Kb,
                        __bf16* __restrict__ P, int nTiles) {
  int K = Ka + Kb, KC = K / 32;
  int idx = blockIdx.x * 256 + threadIdx.x;
  int total = nTiles * KC * 512;
  if (idx >= total) return;
  int j = idx & 15, lane = (idx >> 4) & 31;
  int kc = (idx >> 9) % KC, nt = idx / (KC * 512);
  int n = nt * 16 + (lane & 15);
  int k = kc * 32 + ((lane & 16) >> 1) + (j & 7) + ((j & 8) << 1);
  float v = (k < Ka) ? Wa[n * Ka + k] : Wb[n * Kb + (k - Ka)];
  P[idx] = (__bf16)v;
}

__global__ void pack_one(const float* __restrict__ W, __bf16* __restrict__ P,
                         int Nreal, int K, int nTiles) {
  int KC = K / 32;
  int idx = blockIdx.x * 256 + threadIdx.x;
  int total = nTiles * KC * 512;
  if (idx >= total) return;
  int j = idx & 15, lane = (idx >> 4) & 31;
  int kc = (idx >> 9) % KC, nt = idx / (KC * 512);
  int n = nt * 16 + (lane & 15);
  int k = kc * 32 + ((lane & 16) >> 1) + (j & 7) + ((j & 8) << 1);
  P[idx] = (__bf16)((n < Nreal) ? W[n * K + k] : 0.0f);
}

__global__ void pack_heads(const float* __restrict__ wmu,
                           const float* __restrict__ wstd,
                           const float* __restrict__ wq,
                           __bf16* __restrict__ P) {
  // N = 96 (mu|std|q stacked), K = 256
  int idx = blockIdx.x * 256 + threadIdx.x;
  if (idx >= 6 * 8 * 512) return;
  int j = idx & 15, lane = (idx >> 4) & 31;
  int kc = (idx >> 9) % 8, nt = idx / (8 * 512);
  int n = nt * 16 + (lane & 15);
  int k = kc * 32 + ((lane & 16) >> 1) + (j & 7) + ((j & 8) << 1);
  const float* W = (n < 32) ? wmu : (n < 64) ? wstd : wq;
  P[idx] = (__bf16)W[(n & 31) * 256 + k];
}

// f32 -> bf16 row-major mirror (for embedding tables)
__global__ void to_bf16(const float* __restrict__ src, __bf16* __restrict__ dst, int n) {
  int i = blockIdx.x * 256 + threadIdx.x;
  if (i < n) dst[i] = (__bf16)src[i];
}

// ---------------------------------------------------------------- GRU step
// One wave = 16 batch rows x 16 hidden cols, closing r/z/n gates locally.
// KIN = 64 (enc: [emb]) or 128 (dec: [zcat | emb]); K total = KIN + 256.
// All A-side activations are bf16 mirrors; h kept in f32 for the blend.

template <int KIN>
__global__ void gru_step(const int* __restrict__ x, int t,
                         const __bf16* __restrict__ embB,
                         const __bf16* __restrict__ zcatB,
                         const float* __restrict__ h_in,
                         const __bf16* __restrict__ h_inB,
                         const __bf16* __restrict__ packB,
                         const float* __restrict__ bih,
                         const float* __restrict__ bhh,
                         float* __restrict__ h_out,
                         __bf16* __restrict__ h_outB) {
  constexpr int KT = KIN + 256;
  constexpr int KC = KT / 32;
  int lane = threadIdx.x & 31, wave = threadIdx.x >> 5;
  int wg = blockIdx.x * 8 + wave;         // 0..2047
  int mTile = wg >> 4;                    // 0..127
  int g = wg & 15;                        // hidden col group 0..15
  int mrow = (mTile << 4) + (lane & 15);  // A-matrix row (batch index)
  int tok = x[mrow * T_ + t];
  const __bf16* embRow = embB + tok * E_;
  const __bf16* hRow = h_inB + mrow * H_;
  const __bf16* zRow = (KIN == 128) ? (zcatB + mrow * 64) : nullptr;
  int hi = (lane & 16) ? 8 : 0;

  v8f accR = {}, accZ = {}, accXN = {}, accHN = {};
  const __bf16* PR = packB + (size_t)(g) * KC * 512;
  const __bf16* PZ = packB + (size_t)(16 + g) * KC * 512;
  const __bf16* PN = packB + (size_t)(32 + g) * KC * 512;

  for (int kc = 0; kc < KC; ++kc) {
    int k0 = kc * 32;
    const __bf16* base;
    int rel;
    if (k0 < KIN) {
      if (KIN == 128 && k0 >= 64) { base = embRow; rel = k0 - 64; }
      else if (KIN == 128)        { base = zRow;   rel = k0; }
      else                        { base = embRow; rel = k0; }
    } else { base = hRow; rel = k0 - KIN; }

    v16bf a = loadA(base + rel + hi);

    if (kc + 1 < KC) __builtin_prefetch(PR + (kc + 1) * 512 + lane * 16, 0, 1);

    v16bf bR = *(const v16bf*)(PR + kc * 512 + lane * 16);
    v16bf bZ = *(const v16bf*)(PZ + kc * 512 + lane * 16);
    v16bf bN = *(const v16bf*)(PN + kc * 512 + lane * 16);
    accR = wmma_bf16(a, bR, accR);
    accZ = wmma_bf16(a, bZ, accZ);
    if (k0 < KIN) accXN = wmma_bf16(a, bN, accXN);
    else          accHN = wmma_bf16(a, bN, accHN);
  }

  int col = (g << 4) + (lane & 15);
  float biR = bih[col] + bhh[col];
  float biZ = bih[256 + col] + bhh[256 + col];
  float bXN = bih[512 + col];
  float bHN = bhh[512 + col];
  int rowBase = (mTile << 4) + hi;
#pragma unroll
  for (int i = 0; i < 8; ++i) {
    int b = rowBase + i;
    float r = sig_(accR[i] + biR);
    float z = sig_(accZ[i] + biZ);
    float n = tanhf((accXN[i] + bXN) + r * (accHN[i] + bHN));
    float hp = h_in[b * H_ + col];
    float hv = (1.f - z) * n + z * hp;
    h_out[b * H_ + col] = hv;
    h_outB[b * H_ + col] = (__bf16)hv;
  }
}

// ------------------------------------------------- latent heads (mu/std/toq)
__global__ void latent_heads(const __bf16* __restrict__ hB16,
                             const __bf16* __restrict__ P,
                             const float* __restrict__ bmu,
                             const float* __restrict__ bstd,
                             const float* __restrict__ bq,
                             float* __restrict__ mu, float* __restrict__ toq,
                             float* __restrict__ out) {
  int lane = threadIdx.x & 31, wave = threadIdx.x >> 5;
  int wg = blockIdx.x * 8 + wave;  // 0..767
  int mTile = wg / 6, nt = wg % 6;
  int mrow = mTile * 16 + (lane & 15);
  const __bf16* hRow = hB16 + mrow * H_;
  int hi = (lane & 16) ? 8 : 0;
  v8f acc = {};
  const __bf16* PB = P + (size_t)nt * 8 * 512;
  for (int kc = 0; kc < 8; ++kc) {
    v16bf a = loadA(hRow + kc * 32 + hi);
    v16bf b = *(const v16bf*)(PB + kc * 512 + lane * 16);
    acc = wmma_bf16(a, b, acc);
  }
  int c96 = nt * 16 + (lane & 15);
  int rowBase = mTile * 16 + hi;
  float kl = 0.f;
#pragma unroll
  for (int i = 0; i < 8; ++i) {
    int b = rowBase + i;
    float v = acc[i];
    if (c96 < 32)      { v += bmu[c96];       mu[b * 32 + c96] = v; kl += 0.5f * v * v; }
    else if (c96 < 64) { v += bstd[c96 - 32]; kl += 0.5f * (__expf(v) - v - 1.f); }
    else               { v += bq[c96 - 64];   toq[b * 32 + (c96 - 64)] = v; }
  }
  for (int m = 16; m >= 1; m >>= 1) kl += __shfl_xor(kl, m, 32);
  if (lane == 0) atomicAdd(&out[KL_OFF], kl);
}

// ---------------------------------------------------------------- VQ argmin
__global__ void vq_kernel(const float* __restrict__ toq,
                          const float* __restrict__ mu,
                          const float* __restrict__ cb,
                          __bf16* __restrict__ zcatB, float* __restrict__ out) {
  int b = blockIdx.x * 256 + threadIdx.x;
  if (b >= B_) return;
  float t[32];
#pragma unroll
  for (int j = 0; j < 32; ++j) t[j] = toq[b * 32 + j];
  int best = 0;
  float bd = 3.4e38f;
  for (int k = 0; k < KCB_; ++k) {
    const float* c = cb + k * 32;
    float dot = 0.f, cc = 0.f;
#pragma unroll 8
    for (int j = 0; j < 32; ++j) { float cv = c[j]; dot += cv * t[j]; cc += cv * cv; }
    float d = cc - 2.f * dot;  // |toq|^2 constant per row
    if (d < bd) { bd = d; best = k; }
  }
  const float* c = cb + best * 32;
  float s = 0.f;
#pragma unroll
  for (int j = 0; j < 32; ++j) {
    float df = c[j] - t[j];
    s += df * df;
    zcatB[b * 64 + j] = (__bf16)c[j];                 // q_st forward value == q
    zcatB[b * 64 + 32 + j] = (__bf16)mu[b * 32 + j];  // z == mu
  }
  atomicAdd(&out[Q_OFF], 0.3f * s / 32.f);  // (0.1 + 0.2) * mean
  out[IDX_OFF + b] = (float)best;
}

// --------------------------------------------------------- up-projection
__global__ void up_proj(const __bf16* __restrict__ zcatB,
                        const __bf16* __restrict__ P,
                        const float* __restrict__ bup,
                        float* __restrict__ h, __bf16* __restrict__ hB16) {
  int lane = threadIdx.x & 31, wave = threadIdx.x >> 5;
  int wg = blockIdx.x * 8 + wave;  // 0..2047
  int mTile = wg >> 4, nt = wg & 15;
  int mrow = mTile * 16 + (lane & 15);
  const __bf16* zRow = zcatB + mrow * 64;
  int hi = (lane & 16) ? 8 : 0;
  v8f acc = {};
  const __bf16* PB = P + (size_t)nt * 2 * 512;
#pragma unroll
  for (int kc = 0; kc < 2; ++kc) {
    v16bf a = loadA(zRow + kc * 32 + hi);
    v16bf b = *(const v16bf*)(PB + kc * 512 + lane * 16);
    acc = wmma_bf16(a, b, acc);
  }
  int col = nt * 16 + (lane & 15);
  int rowBase = mTile * 16 + hi;
#pragma unroll
  for (int i = 0; i < 8; ++i) {
    float v = acc[i] + bup[col];
    h[(rowBase + i) * H_ + col] = v;
    hB16[(rowBase + i) * H_ + col] = (__bf16)v;
  }
}

// --------------------------------- decoder vocab head + log-softmax + loss
__global__ void dec_head(const __bf16* __restrict__ hB16,
                         const __bf16* __restrict__ P,
                         const float* __restrict__ bout,
                         const int* __restrict__ x, int t,
                         float* __restrict__ out) {
  __shared__ float lds[8][16][64];
  int lane = threadIdx.x & 31, wave = threadIdx.x >> 5;
  int mTile = blockIdx.x * 8 + wave;  // 0..127
  int mrow = mTile * 16 + (lane & 15);
  const __bf16* hRow = hB16 + mrow * H_;
  int hi = (lane & 16) ? 8 : 0;
  v8f acc[4];
#pragma unroll
  for (int nt = 0; nt < 4; ++nt) acc[nt] = {};
  for (int kc = 0; kc < 8; ++kc) {
    v16bf a = loadA(hRow + kc * 32 + hi);
#pragma unroll
    for (int nt = 0; nt < 4; ++nt) {
      v16bf b = *(const v16bf*)(P + ((size_t)nt * 8 + kc) * 512 + lane * 16);
      acc[nt] = wmma_bf16(a, b, acc[nt]);
    }
  }
  int rb = (lane & 16) ? 8 : 0;
#pragma unroll
  for (int nt = 0; nt < 4; ++nt)
#pragma unroll
    for (int i = 0; i < 8; ++i)
      lds[wave][rb + i][nt * 16 + (lane & 15)] = acc[nt][i];
  __syncthreads();

  float xl = 0.f, cnt = 0.f;
  if (lane < 16) {
    int row = lane;
    int b = mTile * 16 + row;
    int tgt = x[b * T_ + t + 1];
    float mx = -3.4e38f;
    int am = 0;
    for (int c = 0; c < V_; ++c) {
      float v = lds[wave][row][c] + bout[c];
      lds[wave][row][c] = v;
      if (v > mx) { mx = v; am = c; }
    }
    float se = 0.f;
    for (int c = 0; c < V_; ++c) se += __expf(lds[wave][row][c] - mx);
    float logZ = mx + __logf(se);
    out[PRED_OFF + b * (T_ - 1) + t] = (float)am;
    if (tgt != 0) { xl = logZ - lds[wave][row][tgt]; cnt = 1.f; }
  }
  for (int m = 16; m >= 1; m >>= 1) { xl += __shfl_xor(xl, m, 32); cnt += __shfl_xor(cnt, m, 32); }
  if (lane == 0) { atomicAdd(&out[0], xl); atomicAdd(&out[1], cnt); }
}

__global__ void init_out(float* __restrict__ out) {
  int i = threadIdx.x;
  if (i == 0) out[0] = 0.f;
  if (i == 1) out[1] = 0.f;
  if (i == 2) out[KL_OFF] = 0.f;
  if (i == 3) out[Q_OFF] = 0.f;
}

// ------------------------------------------------------------------- launch
extern "C" void kernel_launch(void* const* d_in, const int* in_sizes, int n_in,
                              void* d_out, int out_size, void* d_ws, size_t ws_size,
                              hipStream_t stream) {
  (void)in_sizes; (void)n_in; (void)out_size; (void)ws_size;
  const int*   x        = (const int*)d_in[0];
  const float* enc_emb  = (const float*)d_in[1];
  const float* enc_wih  = (const float*)d_in[2];
  const float* enc_whh  = (const float*)d_in[3];
  const float* enc_bih  = (const float*)d_in[4];
  const float* enc_bhh  = (const float*)d_in[5];
  const float* w_mu     = (const float*)d_in[6];
  const float* b_mu     = (const float*)d_in[7];
  const float* w_std    = (const float*)d_in[8];
  const float* b_std    = (const float*)d_in[9];
  const float* w_q      = (const float*)d_in[10];
  const float* b_q      = (const float*)d_in[11];
  const float* w_up     = (const float*)d_in[12];
  const float* b_up     = (const float*)d_in[13];
  const float* codebook = (const float*)d_in[14];
  const float* dec_emb  = (const float*)d_in[15];
  const float* dec_wih  = (const float*)d_in[16];
  const float* dec_whh  = (const float*)d_in[17];
  const float* dec_bih  = (const float*)d_in[18];
  const float* dec_bhh  = (const float*)d_in[19];
  const float* w_out    = (const float*)d_in[20];
  const float* b_out    = (const float*)d_in[21];
  float* out = (float*)d_out;

  // workspace carve-up (all chunks 256B-aligned)
  char* w = (char*)d_ws;
  __bf16* packEnc   = (__bf16*)w; w += (size_t)320 * 768 * 2;  // 480 KB
  __bf16* packDec   = (__bf16*)w; w += (size_t)384 * 768 * 2;  // 576 KB
  __bf16* packHeads = (__bf16*)w; w += (size_t)256 * 96 * 2;
  __bf16* packUp    = (__bf16*)w; w += (size_t)64 * 256 * 2;
  __bf16* packOut   = (__bf16*)w; w += (size_t)256 * 64 * 2;
  __bf16* encEmbB   = (__bf16*)w; w += (size_t)64 * E_ * 2;    // 51 rows used
  __bf16* decEmbB   = (__bf16*)w; w += (size_t)64 * E_ * 2;
  float*  hA   = (float*)w;  w += (size_t)B_ * H_ * 4;         // 2 MB
  float*  hB   = (float*)w;  w += (size_t)B_ * H_ * 4;         // 2 MB
  __bf16* hAb  = (__bf16*)w; w += (size_t)B_ * H_ * 2;         // 1 MB
  __bf16* hBb  = (__bf16*)w; w += (size_t)B_ * H_ * 2;         // 1 MB
  float*  mu   = (float*)w;  w += (size_t)B_ * 32 * 4;
  float*  toq  = (float*)w;  w += (size_t)B_ * 32 * 4;
  __bf16* zcatB = (__bf16*)w; w += (size_t)B_ * 64 * 2;

  init_out<<<1, 64, 0, stream>>>(out);

  // pack weights into WMMA B-fragment layout (bf16); bf16 embedding mirrors
  { int tot = 48 * 10 * 512;
    pack_b2<<<(tot + 255) / 256, 256, 0, stream>>>(enc_wih, 64, enc_whh, 256, packEnc, 48); }
  { int tot = 48 * 12 * 512;
    pack_b2<<<(tot + 255) / 256, 256, 0, stream>>>(dec_wih, 128, dec_whh, 256, packDec, 48); }
  { int tot = 6 * 8 * 512;
    pack_heads<<<(tot + 255) / 256, 256, 0, stream>>>(w_mu, w_std, w_q, packHeads); }
  { int tot = 16 * 2 * 512;
    pack_one<<<(tot + 255) / 256, 256, 0, stream>>>(w_up, packUp, 256, 64, 16); }
  { int tot = 4 * 8 * 512;
    pack_one<<<(tot + 255) / 256, 256, 0, stream>>>(w_out, packOut, V_, 256, 4); }
  to_bf16<<<(V_ * E_ + 255) / 256, 256, 0, stream>>>(enc_emb, encEmbB, V_ * E_);
  to_bf16<<<(V_ * E_ + 255) / 256, 256, 0, stream>>>(dec_emb, decEmbB, V_ * E_);

  hipMemsetAsync(hA, 0, (size_t)B_ * H_ * 4, stream);
  hipMemsetAsync(hAb, 0, (size_t)B_ * H_ * 2, stream);

  // encoder GRU: 64 steps, ping-pong h (f32 + bf16 mirror)
  float*  hin = hA;   float*  hout = hB;
  __bf16* hinB = hAb; __bf16* houtB = hBb;
  for (int t = 0; t < T_; ++t) {
    gru_step<64><<<256, 256, 0, stream>>>(x, t, encEmbB, nullptr, hin, hinB,
                                          packEnc, enc_bih, enc_bhh, hout, houtB);
    float* tf = hin; hin = hout; hout = tf;
    __bf16* tb = hinB; hinB = houtB; houtB = tb;
  }
  // h_last: f32 in hin, bf16 in hinB (back in hA/hAb after 64 swaps)

  latent_heads<<<96, 256, 0, stream>>>(hinB, packHeads, b_mu, b_std, b_q, mu, toq, out);
  vq_kernel<<<(B_ + 255) / 256, 256, 0, stream>>>(toq, mu, codebook, zcatB, out);
  up_proj<<<256, 256, 0, stream>>>(zcatB, packUp, b_up, hout, houtB);  // hidden

  // decoder GRU: 63 steps, vocab head each step
  { float* tf = hin; hin = hout; hout = tf;
    __bf16* tb = hinB; hinB = houtB; houtB = tb; }  // hin = hidden
  for (int t = 0; t < T_ - 1; ++t) {
    gru_step<128><<<256, 256, 0, stream>>>(x, t, decEmbB, zcatB, hin, hinB,
                                           packDec, dec_bih, dec_bhh, hout, houtB);
    dec_head<<<16, 256, 0, stream>>>(houtB, packOut, b_out, x, t, out);
    float* tf = hin; hin = hout; hout = tf;
    __bf16* tb = hinB; hinB = houtB; houtB = tb;
  }
}
